// MultiEncAttention_17987323035778
// MI455X (gfx1250) — matrix operand order repaired and checked by reading, not compiled, not run on hardware
//
#include <hip/hip_runtime.h>
#include <hip/hip_bf16.h>

typedef __attribute__((ext_vector_type(16))) __bf16 v16bf;
typedef __attribute__((ext_vector_type(8)))  float  v8f;

#define DMODEL 1024
#define NH     8
#define DK     64
#define BATCH  8
#define LQN    1024
#define LKN    1024

// ---- WMMA fragment lane/element -> matrix coordinate helpers (ISA 7.12.2) ----
// A 16x32 bf16: lane&15 = M row; lane>>4 selects K-half; elems 0-7 -> K half*8+j,
// elems 8-15 -> K 16+half*8+(j-8).
// B 32x16 bf16: lane&15 = N col; elems 0-7 -> K (lane>>4)*8+j, elems 8-15 -> K 16+(lane>>4)*8+(j-8).
// C/D 16x16 f32: VGPR r, lane l -> (M = r + 8*(l>>4), N = l&15).

// ---------------- elementwise fp32 -> bf16 convert (weights) ----------------
__global__ __launch_bounds__(256) void cvt_kernel(const float* __restrict__ in,
                                                  __bf16* __restrict__ out, int n) {
  int i = blockIdx.x * 256 + threadIdx.x;
  if (i < n) out[i] = (__bf16)in[i];
}

// ---------------- LayerNorm: one block per row of q ----------------
__global__ __launch_bounds__(256) void ln_kernel(const float* __restrict__ x,
                                                 const float* __restrict__ g,
                                                 const float* __restrict__ bta,
                                                 __bf16* __restrict__ y) {
  __shared__ float red[256];
  const int row = blockIdx.x;
  const float* xr = x + (size_t)row * DMODEL;
  float s = 0.f;
  for (int i = threadIdx.x; i < DMODEL; i += 256) s += xr[i];
  red[threadIdx.x] = s; __syncthreads();
  for (int o = 128; o > 0; o >>= 1) {
    if ((int)threadIdx.x < o) red[threadIdx.x] += red[threadIdx.x + o];
    __syncthreads();
  }
  const float mu = red[0] * (1.f / DMODEL);
  __syncthreads();
  float v = 0.f;
  for (int i = threadIdx.x; i < DMODEL; i += 256) { float d = xr[i] - mu; v += d * d; }
  red[threadIdx.x] = v; __syncthreads();
  for (int o = 128; o > 0; o >>= 1) {
    if ((int)threadIdx.x < o) red[threadIdx.x] += red[threadIdx.x + o];
    __syncthreads();
  }
  const float rstd = rsqrtf(red[0] * (1.f / DMODEL) + 1e-5f);
  __bf16* yr = y + (size_t)row * DMODEL;
  for (int i = threadIdx.x; i < DMODEL; i += 256)
    yr[i] = (__bf16)((xr[i] - mu) * rstd * g[i] + bta[i]);
}

// ---------------- LDS-tiled GEMM: block computes 128x64, wave computes 32x32 --------
// EPI 0: scatter bf16 to head-major (B,H,L,64). EPI 1: fp32 + bias + residual.
#define BLK_M 128
#define BLK_N 64
#define APAD  40   // A LDS row stride (bf16): 80B -> 16B-aligned runs, conflict-free
#define BPAD  40   // Bs LDS k stride (bf16), Bs stored transposed [n][k]

template <bool ABF, int EPI>
__global__ __launch_bounds__(256) void gemm_kernel(const void* __restrict__ Ap,
                                                   const __bf16* __restrict__ W,
                                                   void* __restrict__ outp, int N,
                                                   const float* __restrict__ bias,
                                                   const float* __restrict__ resid) {
  __shared__ __bf16 As[BLK_M * APAD];   // [row][k], k contiguous
  __shared__ __bf16 Bs[BLK_N * BPAD];   // [col][k], k contiguous (transposed)
  const int t = threadIdx.x;
  const int lane = t & 31;
  const int wave = t >> 5;
  const int m0 = blockIdx.y * BLK_M;
  const int n0 = blockIdx.x * BLK_N;
  const int mw = (wave >> 1) * 32;      // wave sub-tile offsets inside block tile
  const int nw = (wave & 1) * 32;
  const float*  Af = (const float*)Ap;
  const __bf16* Ab = (const __bf16*)Ap;
  const int c  = lane & 15;
  const int rh = lane >> 4;

  v8f acc[2][2] = {};
  for (int k0 = 0; k0 < DMODEL; k0 += 32) {
    __syncthreads();                    // protect LDS from previous iteration readers
    // stage A tile: 128 rows x 32 k (convert to bf16)
    {
      const int ar = t >> 3;            // 0..31
      const int ak = (t & 7) * 4;       // 0..28
#pragma unroll
      for (int p = 0; p < 4; ++p) {
        const int row = p * 32 + ar;
        const size_t gi = (size_t)(m0 + row) * DMODEL + (k0 + ak);
#pragma unroll
        for (int i = 0; i < 4; ++i) {
          if constexpr (ABF) As[row * APAD + ak + i] = Ab[gi + i];
          else               As[row * APAD + ak + i] = (__bf16)Af[gi + i];
        }
      }
      if (!ABF && k0 + 32 < DMODEL)     // prefetch next A k-slice
        __builtin_prefetch((const void*)(Af + (size_t)(m0 + ar) * DMODEL + k0 + 32), 0, 0);
    }
    // stage B tile transposed: Bs[col][k] from W row-major (K x N)
    {
      const int kr = t >> 3;            // 0..31
      const int c0 = (t & 7) * 8;       // 0..56
      const __bf16* wp = W + (size_t)(k0 + kr) * N + n0 + c0;
#pragma unroll
      for (int i = 0; i < 8; ++i) Bs[(c0 + i) * BPAD + kr] = wp[i];
    }
    __syncthreads();

    v16bf a[2], b[2];
#pragma unroll
    for (int s = 0; s < 2; ++s) {
      const int abase = (mw + s * 16 + c) * APAD + rh * 8;
#pragma unroll
      for (int j = 0; j < 8; ++j) a[s][j] = As[abase + j];
#pragma unroll
      for (int j = 0; j < 8; ++j) a[s][8 + j] = As[abase + 16 + j];
      const int bbase = (nw + s * 16 + c) * BPAD + rh * 8;
#pragma unroll
      for (int j = 0; j < 16; ++j) b[s][j] = Bs[bbase + j + (j & 8)];
    }
#pragma unroll
    for (int si = 0; si < 2; ++si)
#pragma unroll
      for (int sj = 0; sj < 2; ++sj)
        acc[si][sj] = __builtin_amdgcn_wmma_f32_16x16x32_bf16(
            false, a[si], false, b[sj], (short)0, acc[si][sj], false, false);
  }

#pragma unroll
  for (int si = 0; si < 2; ++si)
#pragma unroll
    for (int sj = 0; sj < 2; ++sj)
#pragma unroll
      for (int r = 0; r < 8; ++r) {
        const int row = m0 + mw + si * 16 + r + rh * 8;
        const int col = n0 + nw + sj * 16 + c;
        if constexpr (EPI == 0) {
          __bf16* out = (__bf16*)outp;
          const int b_ = row >> 10, l = row & 1023, h = col >> 6, d = col & 63;
          out[(((size_t)(b_ * NH + h) << 10) + l) * DK + d] = (__bf16)acc[si][sj][r];
        } else {
          float* out = (float*)outp;
          out[(size_t)row * DMODEL + col] =
              acc[si][sj][r] + bias[col] + resid[(size_t)row * DMODEL + col];
        }
      }
}

// ---------------- scores: block = 16 q-rows x 128 k-cols, Q tile shared in LDS -------
__global__ __launch_bounds__(256) void scores_kernel(const __bf16* __restrict__ Q1,
                                                     const __bf16* __restrict__ K1,
                                                     const __bf16* __restrict__ Q2,
                                                     const __bf16* __restrict__ K2,
                                                     const int* __restrict__ m1,
                                                     const int* __restrict__ m2,
                                                     float* __restrict__ attn) {
  __shared__ __bf16 Qs[16 * 72];
  const int t = threadIdx.x;
  const int lane = t & 31;
  const int wave = t >> 5;
  const int q0 = blockIdx.y * 16;
  const int n0 = blockIdx.x * 128 + wave * 16;
  const int z  = blockIdx.z;            // branch*64 + b*8 + h
  const int branch = z >> 6;
  const int bh = z & 63;
  const int b = bh >> 3;
  const __bf16* Q  = (branch ? Q2 : Q1) + (size_t)bh * LQN * DK;
  const __bf16* Kh = (branch ? K2 : K1) + (size_t)bh * LKN * DK;
  const int* mask  = (branch ? m2 : m1) + (size_t)b * LQN * LKN;

  {  // stage Q tile 16x64
    const int row = t >> 4, c0 = (t & 15) * 4;
    const __bf16* qp = Q + (size_t)(q0 + row) * DK + c0;
#pragma unroll
    for (int i = 0; i < 4; ++i) Qs[row * 72 + c0 + i] = qp[i];
  }
  __syncthreads();

  const int c  = lane & 15;
  const int rh = lane >> 4;
  v8f acc = {};
#pragma unroll
  for (int k0 = 0; k0 < DK; k0 += 32) {
    v16bf a, bw;
    const int abase = c * 72 + k0 + rh * 8;
#pragma unroll
    for (int j = 0; j < 8; ++j) a[j] = Qs[abase + j];
#pragma unroll
    for (int j = 0; j < 8; ++j) a[8 + j] = Qs[abase + 16 + j];
    const __bf16* kp = Kh + (size_t)(n0 + c) * DK + k0 + rh * 8;
#pragma unroll
    for (int j = 0; j < 16; ++j) bw[j] = kp[j + (j & 8)];   // K^T: table k order
    acc = __builtin_amdgcn_wmma_f32_16x16x32_bf16(false, a, false, bw, (short)0, acc,
                                                  false, false);
  }
  float* ap = attn + (size_t)z * LQN * LKN;
#pragma unroll
  for (int r = 0; r < 8; ++r) {
    const int row = q0 + r + rh * 8;
    const int col = n0 + c;
    float s = acc[r] * 0.125f;                    // / sqrt(64)
    if (mask[(size_t)row * LKN + col] == 0) s = -1e9f;
    ap[(size_t)row * LKN + col] = s;
  }
}

// ---------------- softmax in place over rows of length 1024 ----------------
__global__ __launch_bounds__(256) void softmax_kernel(float* __restrict__ attn) {
  __shared__ float red[256];
  float* row = attn + (size_t)blockIdx.x * LKN;
  float v[4];
  float mx = -3.4e38f;
#pragma unroll
  for (int i = 0; i < 4; ++i) { v[i] = row[threadIdx.x + i * 256]; mx = fmaxf(mx, v[i]); }
  red[threadIdx.x] = mx; __syncthreads();
  for (int o = 128; o > 0; o >>= 1) {
    if ((int)threadIdx.x < o) red[threadIdx.x] = fmaxf(red[threadIdx.x], red[threadIdx.x + o]);
    __syncthreads();
  }
  mx = red[0]; __syncthreads();
  float s = 0.f;
#pragma unroll
  for (int i = 0; i < 4; ++i) { v[i] = __expf(v[i] - mx); s += v[i]; }
  red[threadIdx.x] = s; __syncthreads();
  for (int o = 128; o > 0; o >>= 1) {
    if ((int)threadIdx.x < o) red[threadIdx.x] += red[threadIdx.x + o];
    __syncthreads();
  }
  const float inv = 1.f / red[0];
#pragma unroll
  for (int i = 0; i < 4; ++i) row[threadIdx.x + i * 256] = v[i] * inv;
}

// ---------------- O = P @ V : V k-slice staged via async global->LDS copy ------------
// Block = 8 waves x 16 q-rows = 128 rows; V slice (32x64 bf16 = 4KB) shared by all.
__global__ __launch_bounds__(256) void av_kernel(const float* __restrict__ attn,
                                                 const __bf16* __restrict__ V1,
                                                 const __bf16* __restrict__ V2,
                                                 __bf16* __restrict__ Oc) {
  __shared__ __align__(16) __bf16 Vs[32 * 64];   // [k][n], 4096 bytes
  const int t = threadIdx.x;
  const int lane = t & 31;
  const int wave = t >> 5;
  const int q0 = blockIdx.x * 128 + wave * 16;
  const int z  = blockIdx.y;            // branch*64 + b*8 + h
  const int branch = z >> 6;
  const int bh = z & 63;
  const int b = bh >> 3, h = bh & 7;
  const float*  P = attn + (size_t)z * LQN * LKN;
  const __bf16* V = (branch ? V2 : V1) + (size_t)bh * LKN * DK;

  const unsigned lds_dst = ((unsigned)(size_t)(&Vs[0])) + (unsigned)t * 16u;
  const int c  = lane & 15;
  const int rh = lane >> 4;
  v8f acc[4] = {};
  for (int k0 = 0; k0 < LKN; k0 += 32) {
    __syncthreads();
    {  // async copy of the 4KB V slice: one b128 per thread (ASYNCcnt path)
      const unsigned long long g =
          (unsigned long long)(size_t)(V + (size_t)k0 * DK) + (unsigned long long)t * 16ull;
      asm volatile("global_load_async_to_lds_b128 %0, %1, off"
                   :: "v"(lds_dst), "v"(g) : "memory");
      asm volatile("s_wait_asynccnt 0x0" ::: "memory");
    }
    __syncthreads();

    v16bf a;
    const float* pp = P + (size_t)(q0 + c) * LKN + k0 + rh * 8;
#pragma unroll
    for (int j = 0; j < 8; ++j) a[j] = (__bf16)pp[j];
#pragma unroll
    for (int j = 0; j < 8; ++j) a[8 + j] = (__bf16)pp[16 + j];
#pragma unroll
    for (int nt = 0; nt < 4; ++nt) {
      v16bf bw;
#pragma unroll
      for (int j = 0; j < 16; ++j) bw[j] = Vs[(rh * 8 + j + (j & 8)) * 64 + nt * 16 + c];
      acc[nt] = __builtin_amdgcn_wmma_f32_16x16x32_bf16(false, a, false, bw, (short)0,
                                                        acc[nt], false, false);
    }
  }
#pragma unroll
  for (int nt = 0; nt < 4; ++nt)
#pragma unroll
    for (int r = 0; r < 8; ++r) {
      const int qrow = q0 + r + rh * 8;
      const int col  = branch * 512 + h * 64 + nt * 16 + c;
      Oc[((size_t)(b * LQN + qrow) << 10) + col] = (__bf16)acc[nt][r];
    }
}

extern "C" void kernel_launch(void* const* d_in, const int* in_sizes, int n_in,
                              void* d_out, int out_size, void* d_ws, size_t ws_size,
                              hipStream_t stream) {
  const float* q      = (const float*)d_in[0];
  const float* k1     = (const float*)d_in[1];
  const float* v1     = (const float*)d_in[2];
  const float* k2     = (const float*)d_in[3];
  const float* v2     = (const float*)d_in[4];
  const int*   txt_m  = (const int*)d_in[5];
  const int*   syn_m  = (const int*)d_in[6];
  const float* w_qs1  = (const float*)d_in[7];
  const float* w_ks1  = (const float*)d_in[8];
  const float* w_vs1  = (const float*)d_in[9];
  const float* w_qs2  = (const float*)d_in[10];
  const float* w_ks2  = (const float*)d_in[11];
  const float* w_vs2  = (const float*)d_in[12];
  const float* ln_g   = (const float*)d_in[13];
  const float* ln_b   = (const float*)d_in[14];
  const float* fc_w   = (const float*)d_in[15];
  const float* fc_b   = (const float*)d_in[16];

  char* ws = (char*)d_ws;
  const size_t SZ_QN   = (size_t)BATCH * LQN * DMODEL * 2;
  const size_t SZ_HEAD = (size_t)BATCH * NH * LQN * DK * 2;
  const size_t SZ_PW   = (size_t)DMODEL * (NH * DK) * 2;
  const size_t SZ_FCW  = (size_t)DMODEL * DMODEL * 2;

  __bf16* qn  = (__bf16*)ws;  ws += SZ_QN;
  __bf16* Q1  = (__bf16*)ws;  ws += SZ_HEAD;
  __bf16* K1h = (__bf16*)ws;  ws += SZ_HEAD;
  __bf16* V1h = (__bf16*)ws;  ws += SZ_HEAD;
  __bf16* Q2  = (__bf16*)ws;  ws += SZ_HEAD;
  __bf16* K2h = (__bf16*)ws;  ws += SZ_HEAD;
  __bf16* V2h = (__bf16*)ws;  ws += SZ_HEAD;
  __bf16* Oc  = (__bf16*)ws;  ws += SZ_QN;
  __bf16* wq1 = (__bf16*)ws;  ws += SZ_PW;
  __bf16* wk1 = (__bf16*)ws;  ws += SZ_PW;
  __bf16* wv1 = (__bf16*)ws;  ws += SZ_PW;
  __bf16* wq2 = (__bf16*)ws;  ws += SZ_PW;
  __bf16* wk2 = (__bf16*)ws;  ws += SZ_PW;
  __bf16* wv2 = (__bf16*)ws;  ws += SZ_PW;
  __bf16* wfc = (__bf16*)ws;  ws += SZ_FCW;

  const int npw = DMODEL * NH * DK;
  cvt_kernel<<<(npw + 255) / 256, 256, 0, stream>>>(w_qs1, wq1, npw);
  cvt_kernel<<<(npw + 255) / 256, 256, 0, stream>>>(w_ks1, wk1, npw);
  cvt_kernel<<<(npw + 255) / 256, 256, 0, stream>>>(w_vs1, wv1, npw);
  cvt_kernel<<<(npw + 255) / 256, 256, 0, stream>>>(w_qs2, wq2, npw);
  cvt_kernel<<<(npw + 255) / 256, 256, 0, stream>>>(w_ks2, wk2, npw);
  cvt_kernel<<<(npw + 255) / 256, 256, 0, stream>>>(w_vs2, wv2, npw);
  const int nfc = DMODEL * DMODEL;
  cvt_kernel<<<(nfc + 255) / 256, 256, 0, stream>>>(fc_w, wfc, nfc);

  ln_kernel<<<BATCH * LQN, 256, 0, stream>>>(q, ln_g, ln_b, qn);

  // projections: M=8192, N=512, K=1024 (block tile 128x64)
  dim3 pg(512 / BLK_N, (BATCH * LQN) / BLK_M);
  gemm_kernel<true , 0><<<pg, 256, 0, stream>>>(qn, wq1, Q1,  512, nullptr, nullptr);
  gemm_kernel<true , 0><<<pg, 256, 0, stream>>>(qn, wq2, Q2,  512, nullptr, nullptr);
  gemm_kernel<false, 0><<<pg, 256, 0, stream>>>(k1, wk1, K1h, 512, nullptr, nullptr);
  gemm_kernel<false, 0><<<pg, 256, 0, stream>>>(v1, wv1, V1h, 512, nullptr, nullptr);
  gemm_kernel<false, 0><<<pg, 256, 0, stream>>>(k2, wk2, K2h, 512, nullptr, nullptr);
  gemm_kernel<false, 0><<<pg, 256, 0, stream>>>(v2, wv2, V2h, 512, nullptr, nullptr);

  float* out  = (float*)d_out;
  float* attn = out + (size_t)BATCH * LQN * DMODEL;   // txt then syn, contiguous

  scores_kernel<<<dim3(LKN / 128, LQN / 16, 2 * BATCH * NH), 256, 0, stream>>>(
      Q1, K1h, Q2, K2h, txt_m, syn_m, attn);

  softmax_kernel<<<2 * BATCH * NH * LQN, 256, 0, stream>>>(attn);

  av_kernel<<<dim3(LQN / 128, 2 * BATCH * NH), 256, 0, stream>>>(attn, V1h, V2h, Oc);

  gemm_kernel<true, 1><<<dim3(DMODEL / BLK_N, (BATCH * LQN) / BLK_M), 256, 0, stream>>>(
      Oc, wfc, (void*)out, DMODEL, fc_b, q);
}
